// GATDecoder_29815662969291
// MI455X (gfx1250) — matrix operand order repaired and checked by reading, not compiled, not run on hardware
//
#include <hip/hip_runtime.h>

// ---------------------------------------------------------------------------
// GAT decoder for MI455X (gfx1250): bf16 WMMA GEMMs (4-tile N-blocked) +
// fp32 edge softmax / scatter-aggregate.
// ---------------------------------------------------------------------------

typedef unsigned short bf16_t;
typedef __attribute__((ext_vector_type(16))) __bf16 v16bf;
typedef __attribute__((ext_vector_type(8)))  float  v8f;

#define N_NODES 10000
#define NE      160000
#define BB      4
#define DD      256
#define HH      4
#define DHH     64
#define LL      2
#define HIDD    128
#define NEG_SLOPE 0.1f
#define MROWS   (BB * N_NODES)      // 40000 rows, divisible by 16

__device__ __forceinline__ bf16_t f2bf(float f) {
  unsigned u = __float_as_uint(f);
  u += 0x7FFFu + ((u >> 16) & 1u);      // round to nearest even
  return (bf16_t)(u >> 16);
}
__device__ __forceinline__ float bf2f(bf16_t b) {
  return __uint_as_float(((unsigned)b) << 16);
}

__device__ __forceinline__ void atomicMaxF(float* a, float v) {
  if (v >= 0.f) atomicMax((int*)a, __float_as_int(v));
  else          atomicMin((unsigned int*)a, __float_as_uint(v));
}

union TileU { v16bf v; uint4 q[2]; };

// A-tile 16x32 bf16 (MxK), contiguous k order (matches B):
//  lanes 0-15  (M=row): VGPR0-7 hold K=0..15
//  lanes 16-31 (M=row): VGPR0-7 hold K=16..31
__device__ __forceinline__ v16bf load_a_tile(const bf16_t* A, int lda,
                                             int m0, int k0, int lane) {
  int row = lane & 15, half = lane >> 4;
  const bf16_t* p = A + (size_t)(m0 + row) * lda + k0;
  TileU t;
  t.q[0] = *(const uint4*)(p + half * 16);
  t.q[1] = *(const uint4*)(p + half * 16 + 8);
  return t.v;
}

// B-tile 32x16 bf16 (KxN), B[k][n] = W[n][k] with W row-major (ldw):
//  lanes 0-15 (N=col): VGPR j holds K=2j,2j+1   (K=0..15)
//  lanes 16-31        : K=16..31
__device__ __forceinline__ v16bf load_b_tile(const bf16_t* W, int ldw,
                                             int n0, int k0, int lane) {
  int col = lane & 15, half = lane >> 4;
  const bf16_t* p = W + (size_t)(n0 + col) * ldw + k0 + half * 16;
  TileU t;
  t.q[0] = *(const uint4*)(p);
  t.q[1] = *(const uint4*)(p + 8);
  return t.v;
}

#define WMMA_BF16(a, b, c) \
  __builtin_amdgcn_wmma_f32_16x16x32_bf16(false, (a), false, (b), (short)0, (c), false, false)

// ---------------------------------------------------------------------------
// Elementwise / setup kernels
// ---------------------------------------------------------------------------
__global__ void k_f32_to_bf16(const float* __restrict__ in,
                              bf16_t* __restrict__ out, int n) {
  int i = blockIdx.x * blockDim.x + threadIdx.x;
  if (i < n) out[i] = f2bf(in[i]);
}

// x[b,n, back*128+c] = solutions[back,b,n,c]  -> bf16 [MROWS, 256]
__global__ void k_build_x(const float* __restrict__ sol, bf16_t* __restrict__ xbf) {
  int row = blockIdx.x;                       // b*N + n
  int col = threadIdx.x;                      // 0..255
  int b = row / N_NODES, n = row % N_NODES;
  int back = col >> 7, c = col & 127;
  float v = sol[(((size_t)back * BB + b) * N_NODES + n) * HIDD + c];
  xbf[(size_t)row * DD + col] = f2bf(v);
}

__global__ void k_init(float* __restrict__ rst, float* __restrict__ mb,
                       float* __restrict__ sb) {
  int i = blockIdx.x * blockDim.x + threadIdx.x;
  if (i < MROWS * DD) rst[i] = 0.f;
  if (i < MROWS * HH) { mb[i] = -__builtin_inff(); sb[i] = 0.f; }
}

// ---------------------------------------------------------------------------
// GEMM: feat = X @ fc_w^T  (M=40000, N=256, K=256), bf16 in/out.
// Each wave computes a 16x64 strip: one A load feeds 4 WMMAs.
// grid 2500 x 128 threads (4 waves) -> 10000 waves = 2500 M-tiles x 4 strips.
// ---------------------------------------------------------------------------
__global__ void k_gemm_feat(const bf16_t* __restrict__ A, int lda,
                            const bf16_t* __restrict__ W,
                            bf16_t* __restrict__ featbf) {
  int wave = blockIdx.x * 4 + (threadIdx.x >> 5);
  int lane = threadIdx.x & 31;
  int m0  = (wave >> 2) * 16;
  int nb  = (wave & 3) * 64;
  v8f c0 = {}, c1 = {}, c2 = {}, c3 = {};
  for (int k0 = 0; k0 < DD; k0 += 32) {
    v16bf a = load_a_tile(A, lda, m0, k0, lane);
    c0 = WMMA_BF16(a, load_b_tile(W, DD, nb +  0, k0, lane), c0);
    c1 = WMMA_BF16(a, load_b_tile(W, DD, nb + 16, k0, lane), c1);
    c2 = WMMA_BF16(a, load_b_tile(W, DD, nb + 32, k0, lane), c2);
    c3 = WMMA_BF16(a, load_b_tile(W, DD, nb + 48, k0, lane), c3);
  }
  int ncol  = lane & 15;
  int mbase = m0 + (lane >> 4) * 8;
  for (int r = 0; r < 8; ++r) {
    size_t rowo = (size_t)(mbase + r) * DD + nb + ncol;
    featbf[rowo +  0] = f2bf(c0[r]);
    featbf[rowo + 16] = f2bf(c1[r]);
    featbf[rowo + 32] = f2bf(c2[r]);
    featbf[rowo + 48] = f2bf(c3[r]);
  }
}

// el/er: per (b,n,h) dot of feat-head slice with attn vectors
__global__ void k_elr(const bf16_t* __restrict__ featbf,
                      const float* __restrict__ al, const float* __restrict__ ar,
                      float* __restrict__ el, float* __restrict__ er) {
  int idx = blockIdx.x * blockDim.x + threadIdx.x;   // MROWS*HH
  if (idx >= MROWS * HH) return;
  int row = idx / HH, h = idx % HH;
  const bf16_t* f = featbf + (size_t)row * DD + h * DHH;
  float sl = 0.f, sr = 0.f;
  for (int d = 0; d < DHH; ++d) {
    float fv = bf2f(f[d]);
    sl += fv * al[h * DHH + d];
    sr += fv * ar[h * DHH + d];
  }
  el[idx] = sl;
  er[idx] = sr;
}

// ---------------------------------------------------------------------------
// Edge softmax (3 passes over [E,B,H]) + weighted scatter-aggregate
// ---------------------------------------------------------------------------
__global__ void k_edge_score(const float* __restrict__ el, const float* __restrict__ er,
                             const float* __restrict__ w,
                             const int* __restrict__ src, const int* __restrict__ dst,
                             float* __restrict__ ebuf, float* __restrict__ mb) {
  int idx = blockIdx.x * blockDim.x + threadIdx.x;
  if (idx >= NE * BB * HH) return;
  int e = idx / (BB * HH), r = idx % (BB * HH);
  int b = r / HH, h = r % HH;
  int sn = src[e], dn = dst[e];
  float v = el[(b * N_NODES + sn) * HH + h] + er[(b * N_NODES + dn) * HH + h];
  v = v > 0.f ? v : NEG_SLOPE * v;
  v *= w[e];
  ebuf[idx] = v;
  atomicMaxF(&mb[(b * N_NODES + dn) * HH + h], v);
}

__global__ void k_edge_exp(float* __restrict__ ebuf, const float* __restrict__ mb,
                           float* __restrict__ sb, const int* __restrict__ dst) {
  int idx = blockIdx.x * blockDim.x + threadIdx.x;
  if (idx >= NE * BB * HH) return;
  int e = idx / (BB * HH), r = idx % (BB * HH);
  int b = r / HH, h = r % HH;
  int dn = dst[e];
  float ex = __expf(ebuf[idx] - mb[(b * N_NODES + dn) * HH + h]);
  ebuf[idx] = ex;
  atomicAdd(&sb[(b * N_NODES + dn) * HH + h], ex);
}

// one block per edge; 256 threads = (h, dh); loop over batches.
// a = ex / s[dst] computed at use (norm pass fused away).
__global__ void k_scatter(const bf16_t* __restrict__ featbf,
                          const float* __restrict__ ebuf,
                          const float* __restrict__ sb,
                          const int* __restrict__ src, const int* __restrict__ dst,
                          float* __restrict__ rst) {
  int e = blockIdx.x;
  int t = threadIdx.x;                 // 0..255 -> h = t>>6
  int h = t >> 6;
  int sn = src[e], dn = dst[e];
  for (int b = 0; b < BB; ++b) {
    float a  = ebuf[((size_t)e * BB + b) * HH + h] /
               sb[(b * N_NODES + dn) * HH + h];
    float fv = bf2f(featbf[((size_t)(b * N_NODES + sn)) * DD + t]);
    atomicAdd(&rst[((size_t)(b * N_NODES + dn)) * DD + t], a * fv);
  }
}

// ---------------------------------------------------------------------------
// per-head out_fc GEMM (M=40000, N=64, K=64) + bias + ReLU -> hcat (bf16).
// One wave = (m-tile, head); the 4-tile strip covers the whole head width.
// ---------------------------------------------------------------------------
__global__ void k_gemm_outfc(const bf16_t* __restrict__ rstbf,
                             const bf16_t* __restrict__ Wofc,   // [64,64]
                             const float* __restrict__ bias,    // [64]
                             bf16_t* __restrict__ hcat, int colOff) {
  int wave = blockIdx.x * 4 + (threadIdx.x >> 5);   // 10000 waves
  int lane = threadIdx.x & 31;
  int m0 = (wave >> 2) * 16;
  int h  = wave & 3;
  const bf16_t* A = rstbf + h * DHH;                // head column offset
  v8f c0 = {}, c1 = {}, c2 = {}, c3 = {};
  for (int k0 = 0; k0 < DHH; k0 += 32) {
    v16bf a = load_a_tile(A, DD, m0, k0, lane);
    c0 = WMMA_BF16(a, load_b_tile(Wofc, DHH,  0, k0, lane), c0);
    c1 = WMMA_BF16(a, load_b_tile(Wofc, DHH, 16, k0, lane), c1);
    c2 = WMMA_BF16(a, load_b_tile(Wofc, DHH, 32, k0, lane), c2);
    c3 = WMMA_BF16(a, load_b_tile(Wofc, DHH, 48, k0, lane), c3);
  }
  int ncol  = lane & 15;
  int mbase = m0 + (lane >> 4) * 8;
  float b0 = bias[ncol], b1 = bias[ncol + 16], b2 = bias[ncol + 32], b3 = bias[ncol + 48];
  for (int r = 0; r < 8; ++r) {
    size_t rowo = (size_t)(mbase + r) * (2 * DD) + colOff + h * DHH + ncol;
    float v0 = c0[r] + b0, v1 = c1[r] + b1, v2 = c2[r] + b2, v3 = c3[r] + b3;
    hcat[rowo +  0] = f2bf(v0 > 0.f ? v0 : 0.f);
    hcat[rowo + 16] = f2bf(v1 > 0.f ? v1 : 0.f);
    hcat[rowo + 32] = f2bf(v2 > 0.f ? v2 : 0.f);
    hcat[rowo + 48] = f2bf(v3 > 0.f ? v3 : 0.f);
  }
}

// ---------------------------------------------------------------------------
// final MLP: out = hcat @ mlp_w^T + mlp_b  (M=40000, N=256, K=512), f32 out
// ---------------------------------------------------------------------------
__global__ void k_gemm_mlp(const bf16_t* __restrict__ hcat,
                           const bf16_t* __restrict__ Wm,   // [256,512]
                           const float* __restrict__ bias,
                           float* __restrict__ out) {
  int wave = blockIdx.x * 4 + (threadIdx.x >> 5);
  int lane = threadIdx.x & 31;
  int m0 = (wave >> 2) * 16;
  int nb = (wave & 3) * 64;
  v8f c0 = {}, c1 = {}, c2 = {}, c3 = {};
  for (int k0 = 0; k0 < 2 * DD; k0 += 32) {
    v16bf a = load_a_tile(hcat, 2 * DD, m0, k0, lane);
    c0 = WMMA_BF16(a, load_b_tile(Wm, 2 * DD, nb +  0, k0, lane), c0);
    c1 = WMMA_BF16(a, load_b_tile(Wm, 2 * DD, nb + 16, k0, lane), c1);
    c2 = WMMA_BF16(a, load_b_tile(Wm, 2 * DD, nb + 32, k0, lane), c2);
    c3 = WMMA_BF16(a, load_b_tile(Wm, 2 * DD, nb + 48, k0, lane), c3);
  }
  int ncol  = lane & 15;
  int mbase = m0 + (lane >> 4) * 8;
  float b0 = bias[nb + ncol],      b1 = bias[nb + ncol + 16];
  float b2 = bias[nb + ncol + 32], b3 = bias[nb + ncol + 48];
  for (int r = 0; r < 8; ++r) {
    size_t rowo = (size_t)(mbase + r) * DD + nb + ncol;
    out[rowo +  0] = c0[r] + b0;
    out[rowo + 16] = c1[r] + b1;
    out[rowo + 32] = c2[r] + b2;
    out[rowo + 48] = c3[r] + b3;
  }
}

// ---------------------------------------------------------------------------
extern "C" void kernel_launch(void* const* d_in, const int* in_sizes, int n_in,
                              void* d_out, int out_size, void* d_ws, size_t ws_size,
                              hipStream_t stream) {
  (void)in_sizes; (void)n_in; (void)out_size; (void)ws_size;

  const float* sol    = (const float*)d_in[0];   // [2,4,10000,128]
  const float* w      = (const float*)d_in[1];   // [E]
  const float* fc_w   = (const float*)d_in[2];   // [2,256,256]
  const float* attn_l = (const float*)d_in[3];   // [2,4,64]
  const float* attn_r = (const float*)d_in[4];   // [2,4,64]
  const float* ofc_w  = (const float*)d_in[5];   // [2,64,64]
  const float* ofc_b  = (const float*)d_in[6];   // [2,64]
  const float* mlp_w  = (const float*)d_in[7];   // [256,512]
  const float* mlp_b  = (const float*)d_in[8];   // [256]
  const int*   src    = (const int*)d_in[9];     // [E]
  const int*   dst    = (const int*)d_in[10];    // [E]
  float* out = (float*)d_out;

  // bump allocator over workspace (256B aligned slices)
  char* wp = (char*)d_ws;
  auto alloc = [&](size_t bytes) -> void* {
    void* r = (void*)wp;
    wp += (bytes + 255) & ~(size_t)255;
    return r;
  };

  bf16_t* xbf    = (bf16_t*)alloc((size_t)MROWS * DD * 2);        // concat input (bf16)
  bf16_t* fcwbf  = (bf16_t*)alloc((size_t)LL * DD * DD * 2);
  bf16_t* ofcwbf = (bf16_t*)alloc((size_t)LL * DHH * DHH * 2);
  bf16_t* mlpwbf = (bf16_t*)alloc((size_t)DD * 2 * DD * 2);
  bf16_t* featbf = (bf16_t*)alloc((size_t)MROWS * DD * 2);
  float*  el     = (float*)alloc((size_t)MROWS * HH * 4);
  float*  er     = (float*)alloc((size_t)MROWS * HH * 4);
  float*  mbuf   = (float*)alloc((size_t)MROWS * HH * 4);
  float*  sbuf   = (float*)alloc((size_t)MROWS * HH * 4);
  float*  ebuf   = (float*)alloc((size_t)NE * BB * HH * 4);
  float*  rst    = (float*)alloc((size_t)MROWS * DD * 4);
  bf16_t* rstbf  = (bf16_t*)alloc((size_t)MROWS * DD * 2);
  bf16_t* hcat   = (bf16_t*)alloc((size_t)MROWS * 2 * DD * 2);    // [40000, 512]

  // weights -> bf16 (every call; deterministic)
  k_f32_to_bf16<<<(LL * DD * DD + 255) / 256, 256, 0, stream>>>(fc_w, fcwbf, LL * DD * DD);
  k_f32_to_bf16<<<(LL * DHH * DHH + 255) / 256, 256, 0, stream>>>(ofc_w, ofcwbf, LL * DHH * DHH);
  k_f32_to_bf16<<<(DD * 2 * DD + 255) / 256, 256, 0, stream>>>(mlp_w, mlpwbf, DD * 2 * DD);

  // concat history -> x (bf16)
  k_build_x<<<MROWS, 256, 0, stream>>>(sol, xbf);

  const int nEBH = NE * BB * HH;     // 2,560,000
  for (int l = 0; l < LL; ++l) {
    const bf16_t* Ain = (l == 0) ? xbf : hcat;      // layer1 input = hcat[:, :256]
    int lda = (l == 0) ? DD : 2 * DD;

    k_gemm_feat<<<2500, 128, 0, stream>>>(Ain, lda, fcwbf + (size_t)l * DD * DD, featbf);
    k_elr<<<(MROWS * HH + 255) / 256, 256, 0, stream>>>(
        featbf, attn_l + l * HH * DHH, attn_r + l * HH * DHH, el, er);
    k_init<<<(MROWS * DD + 255) / 256, 256, 0, stream>>>(rst, mbuf, sbuf);

    k_edge_score<<<(nEBH + 255) / 256, 256, 0, stream>>>(el, er, w, src, dst, ebuf, mbuf);
    k_edge_exp<<<(nEBH + 255) / 256, 256, 0, stream>>>(ebuf, mbuf, sbuf, dst);
    k_scatter<<<NE, 256, 0, stream>>>(featbf, ebuf, sbuf, src, dst, rst);

    k_f32_to_bf16<<<(MROWS * DD + 255) / 256, 256, 0, stream>>>(rst, rstbf, MROWS * DD);
    k_gemm_outfc<<<2500, 128, 0, stream>>>(rstbf, ofcwbf + (size_t)l * DHH * DHH,
                                           ofc_b + l * DHH, hcat, l * DD);
  }

  k_gemm_mlp<<<2500, 128, 0, stream>>>(hcat, mlpwbf, mlp_b, out);
}
